// SteeringAnglePredictor_47923245088897
// MI455X (gfx1250) — matrix operand order repaired and checked by reading, not compiled, not run on hardware
//
#include <hip/hip_runtime.h>
#include <hip/hip_bf16.h>

typedef __attribute__((ext_vector_type(16))) __bf16 v16bf;
typedef __attribute__((ext_vector_type(8)))  float  v8f;
typedef unsigned short u16;

#define BATCH   8192
#define KFC1    5104
#define KP1     5120      // K padded to multiple of 32 for 16x16x32 WMMA
#define N1      64
#define N2      32
#define SSZ     32
#define NN      19
#define UNF     6
#define LOG2E   1.4426950408889634f

// ---- ws layout (bytes, all 256B aligned) ----
#define ACT_OFF   0ull                       // 8192*5120 u16  = 83,886,080
#define W1_OFF    83886080ull                // 64*5120 u16    =    655,360
#define FC1O_OFF  84541440ull                // 8192*64 u16    =  1,048,576
#define W2_OFF    85590016ull                // 32*64 u16      =      4,096
#define FC2O_OFF  85594112ull                // 8192*32 f32    =  1,048,576
#define WNS_OFF   86642688ull                // 8192*19 f32    =    622,592
#define WDS_OFF   87265280ull                // 8192*19 f32    =    622,592

__device__ __forceinline__ u16 f2bf(float f) {
    union { float f; unsigned u; } v; v.f = f;
    unsigned r = v.u + 0x7FFFu + ((v.u >> 16) & 1u);   // RNE
    return (u16)(r >> 16);
}
__device__ __forceinline__ float sigmoidf(float x) { return 1.0f / (1.0f + __expf(-x)); }
__device__ __forceinline__ float softplusf(float x) { return logf(1.0f + __expf(x)); }
__device__ __forceinline__ float bcast(float v, int lane) {        // uniform broadcast
    return __builtin_bit_cast(float,
           __builtin_amdgcn_readlane(__builtin_bit_cast(int, v), lane));
}

union Frag16 { v16bf v; uint4 u4[2]; };

// ---------------- conv 2x2 valid + ELU -> bf16 activations (row-padded to 5120) ----
__global__ void conv_elu_k(const float* __restrict__ x, const float* __restrict__ cw,
                           const float* __restrict__ cb, u16* __restrict__ act) {
    long idx = (long)blockIdx.x * blockDim.x + threadIdx.x;
    if (idx >= (long)BATCH * KFC1) return;
    int  k = (int)(idx % KFC1);
    long b = idx / KFC1;
    int o = k / 638, rem = k % 638, h = rem / 29, wp = rem % 29;   // 638 = 22*29
    float acc = cb[o];
#pragma unroll
    for (int c = 0; c < 3; ++c)
#pragma unroll
        for (int kh = 0; kh < 2; ++kh)
#pragma unroll
            for (int kw = 0; kw < 2; ++kw)
                acc = fmaf(x[((b * 3 + c) * 23 + h + kh) * 30 + wp + kw],
                           cw[((o * 3 + c) * 2 + kh) * 2 + kw], acc);
    float e = acc > 0.0f ? acc : (__expf(acc) - 1.0f);             // ELU(alpha=1)
    act[b * KP1 + k] = f2bf(e);
}

__global__ void pad_act_k(u16* __restrict__ act) {
    int idx = blockIdx.x * blockDim.x + threadIdx.x;               // 8192*16
    if (idx >= BATCH * (KP1 - KFC1)) return;
    int b = idx / 16, p = idx % 16;
    act[(long)b * KP1 + KFC1 + p] = 0;
}

__global__ void cvt_w1_k(const float* __restrict__ w, u16* __restrict__ wb) {
    int idx = blockIdx.x * blockDim.x + threadIdx.x;               // 64*5120
    if (idx >= N1 * KP1) return;
    int n = idx / KP1, k = idx % KP1;
    wb[idx] = f2bf(k < KFC1 ? w[n * KFC1 + k] : 0.0f);
}

__global__ void cvt_w2_k(const float* __restrict__ w, u16* __restrict__ wb) {
    int idx = blockIdx.x * blockDim.x + threadIdx.x;               // 32*64
    if (idx >= N2 * N1) return;
    wb[idx] = f2bf(w[idx]);
}

// ---- FC1: (8192x5120 bf16) @ (5120x64 bf16) + b, ReLU -> bf16 ----------------------
// One wave computes a 16x64 strip: A fragment loaded once per k-step, reused across
// 4 N-tiles. Single-stage wide-issue: all 10 b128 loads of the k-step are issued in
// one clause BEFORE the 4 WMMAs, so waits stagger (0x6/0x4/0x2/0x0) and next-step
// loads overlap the last WMMA. ~90 live VGPRs -> no spills.
__global__ void fc1_wmma_k(const u16* __restrict__ A, const u16* __restrict__ Bm,
                           const float* __restrict__ bias, u16* __restrict__ out) {
    int wave = (blockIdx.x * blockDim.x + threadIdx.x) >> 5;       // 0..511 = M-tile
    int lane = threadIdx.x & 31;
    int lo = lane & 15, hf = lane >> 4;
    const u16* pA = A  + (size_t)(wave * 16 + lo) * KP1 + hf * 8;  // A: row = lo
    const u16* pB = Bm + (size_t)lo * KP1 + hf * 8;                // B: col = lo
    v8f acc0 = {0.f,0.f,0.f,0.f,0.f,0.f,0.f,0.f};
    v8f acc1 = acc0, acc2 = acc0, acc3 = acc0;
    for (int k0 = 0; k0 < KP1; k0 += 32) {
        __builtin_prefetch(pA + k0 + 512, 0, 1);                   // global_prefetch_b8
        Frag16 a, b0, b1, b2, b3;                                  // distinct registers
        a.u4[0]  = *reinterpret_cast<const uint4*>(pA + k0);       // K = k0+hf*8 ..
        a.u4[1]  = *reinterpret_cast<const uint4*>(pA + k0 + 16);  // K = k0+16+hf*8 ..
        b0.u4[0] = *reinterpret_cast<const uint4*>(pB + k0);       // K = k0+hf*8 ..
        b0.u4[1] = *reinterpret_cast<const uint4*>(pB + k0 + 16);
        b1.u4[0] = *reinterpret_cast<const uint4*>(pB + 81920 + k0);
        b1.u4[1] = *reinterpret_cast<const uint4*>(pB + 81920 + k0 + 16);
        b2.u4[0] = *reinterpret_cast<const uint4*>(pB + 163840 + k0);
        b2.u4[1] = *reinterpret_cast<const uint4*>(pB + 163840 + k0 + 16);
        b3.u4[0] = *reinterpret_cast<const uint4*>(pB + 245760 + k0);
        b3.u4[1] = *reinterpret_cast<const uint4*>(pB + 245760 + k0 + 16);
        acc0 = __builtin_amdgcn_wmma_f32_16x16x32_bf16(false, a.v, false, b0.v,
                                                       (short)0, acc0, false, false);
        acc1 = __builtin_amdgcn_wmma_f32_16x16x32_bf16(false, a.v, false, b1.v,
                                                       (short)0, acc1, false, false);
        acc2 = __builtin_amdgcn_wmma_f32_16x16x32_bf16(false, a.v, false, b2.v,
                                                       (short)0, acc2, false, false);
        acc3 = __builtin_amdgcn_wmma_f32_16x16x32_bf16(false, a.v, false, b3.v,
                                                       (short)0, acc3, false, false);
    }
    int mbase = wave * 16 + hf * 8;
    v8f accs[4] = {acc0, acc1, acc2, acc3};
#pragma unroll
    for (int j = 0; j < 4; ++j) {
        int n = j * 16 + lo;
        float bn = bias[n];
#pragma unroll
        for (int r = 0; r < 8; ++r) {
            float v = accs[j][r] + bn;
            v = v > 0.0f ? v : 0.0f;                               // ReLU
            out[(size_t)(mbase + r) * N1 + n] = f2bf(v);
        }
    }
}

// ---------------- FC2: (8192x64 bf16) @ (64x32 bf16) + b -> f32 ---------------------
__global__ void fc2_wmma_k(const u16* __restrict__ A, const u16* __restrict__ Bm,
                           const float* __restrict__ bias, float* __restrict__ out) {
    int wave = (blockIdx.x * blockDim.x + threadIdx.x) >> 5;       // 0..1023
    int lane = threadIdx.x & 31;
    int nt = wave & 1;                                             // 2 N-tiles
    int mt = wave >> 1;                                            // 512 M-tiles
    int lo = lane & 15, hf = lane >> 4;
    const u16* pA = A  + (size_t)(mt * 16 + lo) * N1 + hf * 8;
    const u16* pB = Bm + (size_t)(nt * 16 + lo) * N1 + hf * 8;
    v8f acc = {0.f, 0.f, 0.f, 0.f, 0.f, 0.f, 0.f, 0.f};
#pragma unroll
    for (int k0 = 0; k0 < N1; k0 += 32) {
        Frag16 a, b;
        a.u4[0] = *reinterpret_cast<const uint4*>(pA + k0);
        a.u4[1] = *reinterpret_cast<const uint4*>(pA + k0 + 16);
        b.u4[0] = *reinterpret_cast<const uint4*>(pB + k0);
        b.u4[1] = *reinterpret_cast<const uint4*>(pB + k0 + 16);
        acc = __builtin_amdgcn_wmma_f32_16x16x32_bf16(false, a.v, false, b.v,
                                                      (short)0, acc, false, false);
    }
    int n = nt * 16 + lo;
    float bn = bias[n];
    int mbase = mt * 16 + hf * 8;
#pragma unroll
    for (int r = 0; r < 8; ++r)
        out[(size_t)(mbase + r) * N2 + n] = acc[r] + bn;
}

// ---------------- sensory synapse precompute: (T,S) -> wnum_s/wden_s (T,N) ----------
__global__ void sensory_k(const float* __restrict__ y, const float* __restrict__ iw,
                          const float* __restrict__ ib,
                          const float* __restrict__ sw, const float* __restrict__ smu,
                          const float* __restrict__ ssig, const float* __restrict__ ser,
                          const float* __restrict__ smask,
                          float* __restrict__ wns, float* __restrict__ wds) {
    int idx = blockIdx.x * blockDim.x + threadIdx.x;               // t*19+n
    if (idx >= BATCH * NN) return;
    int t = idx / NN, n = idx % NN;
    float wn = 0.0f, wd = 0.0f;
#pragma unroll 4
    for (int s = 0; s < SSZ; ++s) {
        float seq = fmaf(y[t * SSZ + s], iw[s], ib[s]);
        int p = s * NN + n;
        float a = softplusf(sw[p]) * sigmoidf(ssig[p] * (seq - smu[p])) * smask[p];
        wn = fmaf(a, ser[p], wn);
        wd += a;
    }
    wns[idx] = wn;
    wds[idx] = wd;
}

// ---- LTC implicit-Euler scan: single wave32, lane = target neuron ------------------
// Sequential chain is the latency floor: readlane broadcast (no LDS round-trip),
// gate = rcp(1+exp2(fma)), update via rcp -- minimal dependent-op count per unfold.
__global__ void ltc_scan_k(const float* __restrict__ w, const float* __restrict__ mu,
                           const float* __restrict__ sigma, const float* __restrict__ erev,
                           const float* __restrict__ mask, const float* __restrict__ gleak,
                           const float* __restrict__ vleak, const float* __restrict__ cm,
                           const float* __restrict__ wns, const float* __restrict__ wds,
                           const float* __restrict__ ow, const float* __restrict__ ob,
                           float* __restrict__ out) {
    int j  = threadIdx.x;
    int jc = j < NN ? j : NN - 1;          // lanes 19..31 mirror lane 18 (never read)
    float nsg[NN], smc[NN], wp[NN], er[NN];  // register-resident column of params
#pragma unroll
    for (int i = 0; i < NN; ++i) {
        int p = i * NN + jc;               // row i = source, col j = target
        float sg = sigma[p];
        nsg[i] = -sg * LOG2E;              // sigmoid(sg*(v-mu)) = 1/(1+2^(nsg*v+smc))
        smc[i] =  sg * mu[p] * LOG2E;
        wp[i]  = softplusf(w[p]) * mask[p];
        er[i]  = erev[p];
    }
    float cmt  = softplusf(cm[jc]) * (float)UNF;
    float gl   = softplusf(gleak[jc]);
    float glvl = gl * vleak[jc];
    float cge  = cmt + gl + 1e-8f;
    float owv = ow[0], obv = ob[0];
    float v = 0.0f;
    for (int t = 0; t < BATCH; ++t) {
        float wns_t = wns[t * NN + jc];
        float wds_t = wds[t * NN + jc];
#pragma unroll
        for (int u = 0; u < UNF; ++u) {
            float wn = wns_t, wd = wds_t;
#pragma unroll
            for (int i = 0; i < NN; ++i) {
                float vi = bcast(v, i);                            // v_readlane (SALU)
                float e  = __builtin_amdgcn_exp2f(fmaf(nsg[i], vi, smc[i]));
                float gt = __builtin_amdgcn_rcpf(1.0f + e);        // v_rcp_f32
                float wa = wp[i] * gt;
                wn = fmaf(wa, er[i], wn);
                wd += wa;
            }
            v = fmaf(cmt, v, glvl + wn) * __builtin_amdgcn_rcpf(cge + wd);
        }
        if (j == 0) out[t] = fmaf(v, owv, obv);                    // motor readout (M=1)
    }
}

extern "C" void kernel_launch(void* const* d_in, const int* in_sizes, int n_in,
                              void* d_out, int out_size, void* d_ws, size_t ws_size,
                              hipStream_t stream) {
    const float* x      = (const float*)d_in[0];
    const float* conv_w = (const float*)d_in[1];
    const float* conv_b = (const float*)d_in[2];
    const float* fc1_w  = (const float*)d_in[3];
    const float* fc1_b  = (const float*)d_in[4];
    const float* fc2_w  = (const float*)d_in[5];
    const float* fc2_b  = (const float*)d_in[6];
    const float* inp_w  = (const float*)d_in[7];
    const float* inp_b  = (const float*)d_in[8];
    const float* s_w    = (const float*)d_in[9];
    const float* s_mu   = (const float*)d_in[10];
    const float* s_sig  = (const float*)d_in[11];
    const float* s_erev = (const float*)d_in[12];
    const float* s_mask = (const float*)d_in[13];
    const float* w_     = (const float*)d_in[14];
    const float* mu_    = (const float*)d_in[15];
    const float* sig_   = (const float*)d_in[16];
    const float* erev_  = (const float*)d_in[17];
    const float* mask_  = (const float*)d_in[18];
    const float* gleak  = (const float*)d_in[19];
    const float* vleak  = (const float*)d_in[20];
    const float* cm_    = (const float*)d_in[21];
    const float* out_w  = (const float*)d_in[22];
    const float* out_b  = (const float*)d_in[23];
    float* out = (float*)d_out;

    char* ws = (char*)d_ws;
    u16*   actbf = (u16*)  (ws + ACT_OFF);
    u16*   w1b   = (u16*)  (ws + W1_OFF);
    u16*   fc1o  = (u16*)  (ws + FC1O_OFF);
    u16*   w2b   = (u16*)  (ws + W2_OFF);
    float* fc2o  = (float*)(ws + FC2O_OFF);
    float* wns   = (float*)(ws + WNS_OFF);
    float* wds   = (float*)(ws + WDS_OFF);

    // 1) conv + ELU -> bf16 activation matrix (8192 x 5120), pad tail = 0
    conv_elu_k<<<(BATCH * KFC1) / 256, 256, 0, stream>>>(x, conv_w, conv_b, actbf);
    pad_act_k<<<(BATCH * 16) / 256, 256, 0, stream>>>(actbf);
    // 2) weight conversion to bf16 (fc1 padded to K=5120)
    cvt_w1_k<<<(N1 * KP1) / 256, 256, 0, stream>>>(fc1_w, w1b);
    cvt_w2_k<<<(N2 * N1 + 255) / 256, 256, 0, stream>>>(fc2_w, w2b);
    // 3) FC1 GEMM (WMMA bf16, fp32 acc, 16x64 strip per wave, wide-issue loads)
    fc1_wmma_k<<<64, 256, 0, stream>>>(actbf, w1b, fc1_b, fc1o);
    // 4) FC2 GEMM (WMMA) -> f32
    fc2_wmma_k<<<128, 256, 0, stream>>>(fc1o, w2b, fc2_b, fc2o);
    // 5) sensory synapse sums (parallel over T)
    sensory_k<<<(BATCH * NN + 255) / 256, 256, 0, stream>>>(
        fc2o, inp_w, inp_b, s_w, s_mu, s_sig, s_erev, s_mask, wns, wds);
    // 6) sequential LTC scan on one wave32
    ltc_scan_k<<<1, 32, 0, stream>>>(w_, mu_, sig_, erev_, mask_, gleak, vleak, cm_,
                                     wns, wds, out_w, out_b, out);
}
